// BiLSTM_CRF_55868934586453
// MI455X (gfx1250) — compile-verified, matrix operand order checked by reading, not run on hardware
//
#include <hip/hip_runtime.h>
#include <math.h>

// ---------------- types for WMMA ----------------
typedef __attribute__((ext_vector_type(16))) _Float16 v16h;
typedef __attribute__((ext_vector_type(8)))  _Float16 v8h;
typedef __attribute__((ext_vector_type(8)))  float    v8f;

// Problem constants
#define Bn   32
#define Tn   256
#define En   300
#define EP   320           // E padded to multiple of 32
#define H2n  256
#define G4   1024          // 4*H2
#define Kn   12
#define START_T 10
#define STOP_T  11

union FragU { v16h v; v8h h[2]; };

// Load a 16x32 f16 A/B fragment. Lane l holds row/col (l&15);
// halves[0..7]  = [kBase + (l>>4)*8 .. +7]
// halves[8..15] = [kBase + (l>>4)*8 + 16 .. +23]
__device__ inline v16h load_frag(const _Float16* p) {
    FragU u;
    u.h[0] = *(const v8h*)(p);
    u.h[1] = *(const v8h*)(p + 16);
    return u.v;
}

__device__ inline float sigf(float x) { return 1.0f / (1.0f + __expf(-x)); }

// ---------------- prep: embeddings -> f16 (padded), weights -> f16, bias fold ----
__global__ void prep_kernel(const int* __restrict__ sent, const float* __restrict__ emb,
                            const float* __restrict__ WihF, const float* __restrict__ WihB,
                            const float* __restrict__ WhhF, const float* __restrict__ WhhB,
                            const float* __restrict__ bihF, const float* __restrict__ bhhF,
                            const float* __restrict__ bihB, const float* __restrict__ bhhB,
                            _Float16* __restrict__ xh, _Float16* __restrict__ WihH,
                            _Float16* __restrict__ WhhH, float* __restrict__ biasF)
{
    long long idx = (long long)blockIdx.x * blockDim.x + threadIdx.x;
    const long long N_XH  = (long long)Bn * Tn * EP;      // 8192*320
    const long long N_WIH = (long long)G4 * EP;           // 1024*320
    const long long N_WHH = (long long)G4 * H2n;          // 1024*256

    if (idx < N_XH) {
        int row = (int)(idx / EP);
        int e   = (int)(idx % EP);
        float v = 0.0f;
        if (e < En) v = emb[(size_t)sent[row] * En + e];
        xh[idx] = (_Float16)v;
        return;
    }
    idx -= N_XH;
    if (idx < 2 * N_WIH) {
        int dir = (int)(idx / N_WIH);
        long long r = idx % N_WIH;
        int rr = (int)(r / EP);
        int e  = (int)(r % EP);
        const float* W = dir ? WihB : WihF;
        float v = (e < En) ? W[(size_t)rr * En + e] : 0.0f;
        WihH[idx] = (_Float16)v;
        return;
    }
    idx -= 2 * N_WIH;
    if (idx < 2 * N_WHH) {
        int dir = (int)(idx / N_WHH);
        long long r = idx % N_WHH;
        const float* W = dir ? WhhB : WhhF;
        WhhH[idx] = (_Float16)W[r];
        return;
    }
    idx -= 2 * N_WHH;
    if (idx < 2 * G4) {
        int dir = (int)(idx >> 10);
        int j   = (int)(idx & (G4 - 1));
        biasF[idx] = dir ? (bihB[j] + bhhB[j]) : (bihF[j] + bhhF[j]);
    }
}

// ---------------- big GEMM: xg[dir] = xh @ WihH[dir]^T + (bih+bhh) -------------
// M=8192, N=1024, K=320. One wave -> 32(M) x 64(N) tile via 8 WMMA accumulators:
// each B fragment feeds 2 WMMAs, each A fragment feeds 4 WMMAs.
__global__ __launch_bounds__(128) void gemm_xg_kernel(const _Float16* __restrict__ xh,
                                                      const _Float16* __restrict__ WihH,
                                                      const float* __restrict__ biasF,
                                                      float* __restrict__ xg)
{
    int wave = blockIdx.x * (blockDim.x >> 5) + (threadIdx.x >> 5);
    int lane = threadIdx.x & 31;

    int dir = wave >> 12;               // 4096 waves per direction
    int w   = wave & 4095;
    int mtile2 = w >> 4;                // 0..255  (32-row tiles)
    int nblk   = w & 15;                // 0..15 -> 64 columns

    const _Float16* Wd = WihH + (size_t)dir * G4 * EP;
    float* outd        = xg   + (size_t)dir * (Bn * Tn) * G4;
    const float* bias  = biasF + dir * G4;

    v8f zero = {};
    v8f acc[2][4] = { { zero, zero, zero, zero }, { zero, zero, zero, zero } };

    int rowA0 = mtile2 * 32 + (lane & 15);
    int koff  = (lane >> 4) * 8;
    const _Float16* pA0 = xh + (size_t)rowA0 * EP + koff;
    const _Float16* pA1 = pA0 + (size_t)16 * EP;

    for (int kk = 0; kk < EP; kk += 32) {
        __builtin_prefetch(pA0 + kk + 32, 0, 1);
        v16h a0 = load_frag(pA0 + kk);
        v16h a1 = load_frag(pA1 + kk);
#pragma unroll
        for (int n = 0; n < 4; n++) {
            int rowB = nblk * 64 + n * 16 + (lane & 15);
            v16h b = load_frag(Wd + (size_t)rowB * EP + kk + koff);
            acc[0][n] = __builtin_amdgcn_wmma_f32_16x16x32_f16(
                false, a0, false, b, (short)0, acc[0][n], false, false);
            acc[1][n] = __builtin_amdgcn_wmma_f32_16x16x32_f16(
                false, a1, false, b, (short)0, acc[1][n], false, false);
        }
    }

#pragma unroll
    for (int mt = 0; mt < 2; mt++) {
        int m0 = mtile2 * 32 + mt * 16 + (lane >> 4) * 8;
#pragma unroll
        for (int n = 0; n < 4; n++) {
            int col = nblk * 64 + n * 16 + (lane & 15);
            float bcol = bias[col];
#pragma unroll
            for (int j = 0; j < 8; j++) {
                outd[(size_t)(m0 + j) * G4 + col] = acc[mt][n][j] + bcol;
            }
        }
    }
}

// ---------------- persistent recurrent kernel: one block per direction ---------
// h (f16) and c (f32) live in LDS. Each of 32 waves owns a 32(M) x 32(N) tile of
// the 32x1024 gate matrix: g = xg[:,t,:] + h @ Whh^T (biases already folded).
// Every Whh B-tile is loaded exactly once per step across the workgroup.
__global__ __launch_bounds__(1024) void lstm_kernel(const _Float16* __restrict__ WhhH,
                                                    const float* __restrict__ xg,
                                                    const float* __restrict__ h0,
                                                    const float* __restrict__ c0,
                                                    float* __restrict__ gbuf,
                                                    float* __restrict__ hs)
{
    __shared__ _Float16 hL[Bn * H2n];   // 16 KB
    __shared__ float    cL[Bn * H2n];   // 32 KB

    int dir = blockIdx.x;
    const _Float16* Wd = WhhH + (size_t)dir * G4 * H2n;
    const float* xgd   = xg   + (size_t)dir * (Bn * Tn) * G4;
    float* gd          = gbuf + (size_t)dir * Bn * G4;
    float* hsd         = hs   + (size_t)dir * (Bn * Tn) * H2n;

    int tid = threadIdx.x;

    // init h, c from h0/c0
#pragma unroll
    for (int r = 0; r < 8; r++) {
        int idx = tid + r * 1024;      // 0..8191 == b*256 + j
        hL[idx] = (_Float16)h0[dir * (Bn * H2n) + idx];
        cL[idx] = c0[dir * (Bn * H2n) + idx];
    }
    __syncthreads();

    int wave = tid >> 5, lane = tid & 31;   // wave = 32-col block index 0..31
    int koff = (lane >> 4) * 8;
    int rowA0 = (lane & 15);                // batch rows 0..15 / 16..31
    int j8    = (lane >> 4) * 8;

    for (int step = 0; step < Tn; step++) {
        int t = dir ? (Tn - 1 - step) : step;

        // ---- phase 1: g = xg[:,t,:] + h @ Whh^T -------------------------------
        v8f zero = {};
        v8f acc[2][2] = { { zero, zero }, { zero, zero } };
        for (int kk = 0; kk < H2n; kk += 32) {
            v16h a0 = load_frag(&hL[rowA0 * H2n + kk + koff]);          // ds loads
            v16h a1 = load_frag(&hL[(rowA0 + 16) * H2n + kk + koff]);   // ds loads
#pragma unroll
            for (int n = 0; n < 2; n++) {
                int rowB = wave * 32 + n * 16 + (lane & 15);
                v16h b = load_frag(Wd + (size_t)rowB * H2n + kk + koff);
                acc[0][n] = __builtin_amdgcn_wmma_f32_16x16x32_f16(
                    false, a0, false, b, (short)0, acc[0][n], false, false);
                acc[1][n] = __builtin_amdgcn_wmma_f32_16x16x32_f16(
                    false, a1, false, b, (short)0, acc[1][n], false, false);
            }
        }
#pragma unroll
        for (int mt = 0; mt < 2; mt++) {
#pragma unroll
            for (int n = 0; n < 2; n++) {
                int col = wave * 32 + n * 16 + (lane & 15);
#pragma unroll
                for (int j = 0; j < 8; j++) {
                    int b = mt * 16 + j8 + j;
                    gd[(size_t)b * G4 + col] =
                        acc[mt][n][j] + xgd[((size_t)b * Tn + t) * G4 + col];
                }
            }
        }
        __threadfence();
        __syncthreads();

        // ---- phase 2: LSTM cell elementwise -----------------------------------
#pragma unroll
        for (int r = 0; r < 8; r++) {
            int idx = tid + r * 1024;      // b*256 + j
            int b = idx >> 8;
            int j = idx & (H2n - 1);
            float gi = gd[(size_t)b * G4 + j];
            float gf = gd[(size_t)b * G4 + H2n + j];
            float gg = gd[(size_t)b * G4 + 2 * H2n + j];
            float go = gd[(size_t)b * G4 + 3 * H2n + j];
            float c  = sigf(gf) * cL[idx] + sigf(gi) * tanhf(gg);
            float h  = sigf(go) * tanhf(c);
            cL[idx] = c;
            hL[idx] = (_Float16)h;
            hsd[((size_t)b * Tn + t) * H2n + j] = h;
        }
        __syncthreads();
    }
}

// ---------------- feats = [hf|hb] @ Wout^T + bout ------------------------------
__global__ void feats_kernel(const float* __restrict__ hs, const float* __restrict__ Wout,
                             const float* __restrict__ bout, float* __restrict__ feats)
{
    int i = blockIdx.x * blockDim.x + threadIdx.x;      // 0..98303
    if (i >= Bn * Tn * Kn) return;
    int k  = i % Kn;
    int bt = i / Kn;
    const float* hf = hs + (size_t)bt * H2n;
    const float* hb = hs + (size_t)(Bn * Tn) * H2n + (size_t)bt * H2n;
    const float* wf = Wout + (size_t)k * (2 * H2n);
    const float* wb = wf + H2n;
    float s = bout[k];
    for (int j = 0; j < H2n; j += 4) {
        float4 hv = *(const float4*)(hf + j);
        float4 wv = *(const float4*)(wf + j);
        s += hv.x * wv.x + hv.y * wv.y + hv.z * wv.z + hv.w * wv.w;
        float4 hv2 = *(const float4*)(hb + j);
        float4 wv2 = *(const float4*)(wb + j);
        s += hv2.x * wv2.x + hv2.y * wv2.y + hv2.z * wv2.z + hv2.w * wv2.w;
    }
    feats[i] = s;
}

// ---------------- Viterbi: one wave, one thread per batch row -------------------
__global__ void viterbi_kernel(const float* __restrict__ feats, const int* __restrict__ lens,
                               const float* __restrict__ trans, int* __restrict__ bp,
                               float* __restrict__ out)
{
    __shared__ float tr[Kn * Kn];
    for (int i = threadIdx.x; i < Kn * Kn; i += 32) tr[i] = trans[i];
    __syncthreads();

    int b = threadIdx.x;
    if (b >= Bn) return;

    float pre[Kn];
#pragma unroll
    for (int j = 0; j < Kn; j++) pre[j] = -10000.0f;
    pre[START_T] = 0.0f;
    int cl = lens[b];

    for (int t = 0; t < Tn; t++) {
        const float* frow = feats + ((size_t)b * Tn + t) * Kn;
        float nxt[Kn];
#pragma unroll
        for (int j = 0; j < Kn; j++) {
            float best = pre[0] + tr[j * Kn + 0];
            int bi = 0;
#pragma unroll
            for (int i = 1; i < Kn; i++) {
                float v = pre[i] + tr[j * Kn + i];
                if (v > best) { best = v; bi = i; }
            }
            nxt[j] = frow[j] + best;
            bp[((size_t)b * Tn + t) * Kn + j] = bi;
        }
        bool valid = (cl > 0);
        bool last  = (cl == 1);
#pragma unroll
        for (int j = 0; j < Kn; j++) {
            float p = valid ? nxt[j] : pre[j];
            if (last) p += tr[STOP_T * Kn + j];
            pre[j] = p;
        }
        cl -= 1;
    }

    float best = pre[0];
    int idx = 0;
#pragma unroll
    for (int j = 1; j < Kn; j++) {
        if (pre[j] > best) { best = pre[j]; idx = j; }
    }
    out[b] = best;

    float* pout = out + Bn;
    int cur = idx;
    pout[(size_t)b * Tn + (Tn - 1)] = (float)cur;
    for (int t = Tn - 2; t >= 0; t--) {
        cur = bp[((size_t)b * Tn + (t + 1)) * Kn + cur];
        pout[(size_t)b * Tn + t] = (float)cur;
    }
}

// ---------------- host side -----------------------------------------------------
extern "C" void kernel_launch(void* const* d_in, const int* in_sizes, int n_in,
                              void* d_out, int out_size, void* d_ws, size_t ws_size,
                              hipStream_t stream) {
    const int*   sent  = (const int*)d_in[0];
    const int*   lens  = (const int*)d_in[1];
    const float* emb   = (const float*)d_in[2];
    const float* WihF  = (const float*)d_in[3];
    const float* WhhF  = (const float*)d_in[4];
    const float* bihF  = (const float*)d_in[5];
    const float* bhhF  = (const float*)d_in[6];
    const float* WihB  = (const float*)d_in[7];
    const float* WhhB  = (const float*)d_in[8];
    const float* bihB  = (const float*)d_in[9];
    const float* bhhB  = (const float*)d_in[10];
    const float* h0    = (const float*)d_in[11];
    const float* c0    = (const float*)d_in[12];
    const float* Wout  = (const float*)d_in[13];
    const float* bout  = (const float*)d_in[14];
    const float* trans = (const float*)d_in[15];
    float* out = (float*)d_out;

    char* ws = (char*)d_ws;
    size_t off = 0;
    auto alloc = [&](size_t bytes) -> void* {
        void* p = (void*)(ws + off);
        off += (bytes + 255) & ~(size_t)255;
        return p;
    };

    _Float16* xh    = (_Float16*)alloc((size_t)Bn * Tn * EP * 2);        // 5.2 MB
    _Float16* WihH  = (_Float16*)alloc((size_t)2 * G4 * EP * 2);         // 1.3 MB
    _Float16* WhhH  = (_Float16*)alloc((size_t)2 * G4 * H2n * 2);        // 1.0 MB
    float*    biasF = (float*)alloc((size_t)2 * G4 * 4);
    float*    xg    = (float*)alloc((size_t)2 * Bn * Tn * G4 * 4);       // 67 MB
    float*    gbuf  = (float*)alloc((size_t)2 * Bn * G4 * 4);
    float*    hs    = (float*)alloc((size_t)2 * Bn * Tn * H2n * 4);      // 16.8 MB
    float*    feats = (float*)alloc((size_t)Bn * Tn * Kn * 4);
    int*      bp    = (int*)alloc((size_t)Bn * Tn * Kn * 4);

    // 1) prep: gather + f16 conversion + bias fold
    {
        long long total = (long long)Bn * Tn * EP + 2LL * G4 * EP + 2LL * G4 * H2n + 2LL * G4;
        int threads = 256;
        int blocks = (int)((total + threads - 1) / threads);
        prep_kernel<<<blocks, threads, 0, stream>>>(sent, emb, WihF, WihB, WhhF, WhhB,
                                                    bihF, bhhF, bihB, bhhB,
                                                    xh, WihH, WhhH, biasF);
    }

    // 2) bulk WMMA GEMM: xg = xh @ Wih^T + (bih+bhh), both directions
    {
        // 2 dirs * 256 Mtiles(32) * 16 Nblocks(64) = 8192 waves; 4 waves/block
        gemm_xg_kernel<<<2048, 128, 0, stream>>>(xh, WihH, biasF, xg);
    }

    // 3) recurrence: one persistent block per direction
    lstm_kernel<<<2, 1024, 0, stream>>>(WhhH, xg, h0, c0, gbuf, hs);

    // 4) output projection
    {
        int total = Bn * Tn * Kn;
        feats_kernel<<<(total + 255) / 256, 256, 0, stream>>>(hs, Wout, bout, feats);
    }

    // 5) Viterbi decode + backtrack
    viterbi_kernel<<<1, 32, 0, stream>>>(feats, lens, trans, bp, out);
}
